// SelfAttention_27393301414304
// MI455X (gfx1250) — matrix operand order repaired and checked by reading, not compiled, not run on hardware
//
#include <hip/hip_runtime.h>

// ---------------------------------------------------------------------------
// MI455X (gfx1250) self-attention (no softmax) pipeline, bf16 WMMA core.
//   E=512, H=8, N=8, S=1024
// GEMMs: v_wmma_f32_16x16x32_bf16, f32 accumulate.
// Staging: Tensor Data Mover (tensor_load_to_lds, TENSORcnt) for 2D tiles,
//          global_load_async_to_lds_b128 (ASYNCcnt) where per-lane is needed,
//          double-buffered LDS pipeline.
// ---------------------------------------------------------------------------

typedef __bf16 bf16;
typedef unsigned int u32;
typedef unsigned long long u64;
typedef __attribute__((ext_vector_type(8)))  __bf16 v8bf;
typedef __attribute__((ext_vector_type(16))) __bf16 v16bf;
typedef __attribute__((ext_vector_type(8)))  float  v8f;
typedef __attribute__((ext_vector_type(8)))  float  v8ff;
typedef __attribute__((ext_vector_type(4)))  u32    v4u;
typedef __attribute__((ext_vector_type(8)))  u32    v8u;

#define TILE_M 128
#define TILE_N 128
#define TILE_K 32
#define THREADS 256
#define LDS_TILE (TILE_M * TILE_K)     // elements per LDS buffer

// ---- problem dims ----
#define NB   8
#define SEQ  1024
#define EMB  512
#define HDS  8
#define EH   (EMB*HDS)            // 4096
#define MROWS (NB*SEQ)            // 8192

// ---- workspace layout (bytes) ----
#define ACT_BYTES  ((size_t)MROWS*EMB*2)        //  8 MB each (bf16 activations)
#define W_BYTES    ((size_t)EH*EMB*2)           //  4 MB each (bf16 Wv/Wk/Wq)
#define WO_BYTES   ((size_t)EMB*EMB*2)          //  0.5 MB
#define PROJ_BYTES ((size_t)MROWS*EH*2)         // 64 MB each (bf16 Q/K/V)
#define EN_BYTES   ((size_t)NB*HDS*SEQ*SEQ*2)   // 128 MB (bf16 energy)

#define OFF_VX  ((size_t)0)
#define OFF_KX  (OFF_VX + ACT_BYTES)
#define OFF_QX  (OFF_KX + ACT_BYTES)
#define OFF_WV  (OFF_QX + ACT_BYTES)
#define OFF_WK  (OFF_WV + W_BYTES)
#define OFF_WQ  (OFF_WK + W_BYTES)
#define OFF_WO  (OFF_WQ + W_BYTES)
#define OFF_VP  (OFF_WO + WO_BYTES)
#define OFF_KP  (OFF_VP + PROJ_BYTES)
#define OFF_QP  (OFF_KP + PROJ_BYTES)
#define OFF_EN  (OFF_QP + PROJ_BYTES)
#define OFF_AT  (OFF_EN + EN_BYTES)

#define INV_SQRT_E 0.044194173824159216f   // 1/sqrt(512)

// ---------------------------------------------------------------------------
// f32 -> bf16 conversion, 8-wide
// ---------------------------------------------------------------------------
__global__ void __launch_bounds__(256) cvt_f32_bf16(const float* __restrict__ in,
                                                    bf16* __restrict__ out, int n8) {
    int i = blockIdx.x * blockDim.x + threadIdx.x;
    if (i < n8) {
        v8ff v = *(const v8ff*)(in + (size_t)i * 8);
        *(v8bf*)(out + (size_t)i * 8) = __builtin_convertvector(v, v8bf);
    }
}

// ---------------------------------------------------------------------------
// Tensor Data Mover: one descriptor moves a 2D tile (tile_k x tile_rows,
// 2-byte elements, row stride in elements) global -> LDS, packed row-major
// into LDS exactly as sTile[row*TILE_K + col]. Tracked by TENSORcnt.
// D# group0: count=1 | lds_addr | global_addr | type=2 ("image")
// D# group1: data_size=2B, huge tensor dims (tiles are always in-bounds),
//            tile_dim0/1, tensor_dim0_stride. Groups 2/3 unused (2D tensor).
// ---------------------------------------------------------------------------
__device__ __forceinline__ void tdm_load_tile_2d(void* lds_dst, const void* gsrc,
                                                 u32 tile_k, u32 tile_rows,
                                                 u64 row_stride_elems) {
    u64 ga = (u64)(uintptr_t)gsrc;
    v4u g0;
    g0[0] = 1u;                                            // count=1, user mode
    g0[1] = (u32)(uintptr_t)lds_dst;                       // lds_addr
    g0[2] = (u32)ga;                                       // global_addr[31:0]
    g0[3] = (u32)((ga >> 32) & 0x01ffffffu) | (2u << 30);  // addr[56:32] | type=2
    v8u g1;
    g1[0] = (1u << 16);                      // data_size=1 -> 2-byte elements
    g1[1] = 0xffffu << 16;                   // tensor_dim0[15:0]  (0x7fffffff)
    g1[2] = 0x7fffu | (0xffffu << 16);       // tensor_dim0[31:16], tensor_dim1[15:0]
    g1[3] = 0x7fffu | (tile_k << 16);        // tensor_dim1[31:16], tile_dim0
    g1[4] = tile_rows;                       // tile_dim1 (tile_dim2=0 -> 2D tile)
    g1[5] = (u32)row_stride_elems;           // tensor_dim0_stride[31:0]
    g1[6] = (u32)(row_stride_elems >> 32) & 0xffffu;  // stride[47:32]; dim1_stride=0
    g1[7] = 0u;
    asm volatile("tensor_load_to_lds %0, %1" :: "s"(g0), "s"(g1) : "memory");
}
#define WAIT_TENSOR(n) __builtin_amdgcn_s_wait_tensorcnt(n)

// ---------------------------------------------------------------------------
// CDNA5 async global->LDS copy (16B per lane), tracked by ASYNCcnt.
// ---------------------------------------------------------------------------
__device__ __forceinline__ void async_copy_b128(void* lds_dst, const void* gsrc) {
    unsigned lds_off = (unsigned)(uintptr_t)lds_dst;
    unsigned long long gaddr = (unsigned long long)(uintptr_t)gsrc;
    asm volatile("global_load_async_to_lds_b128 %0, %1, off"
                 :: "v"(lds_off), "v"(gaddr)
                 : "memory");
}
#define WAIT_ASYNC(n) asm volatile("s_wait_asynccnt " #n ::: "memory")

// per-lane async staging of a 128x32 bf16 tile (2 ASYNCcnt ops per wave)
__device__ __forceinline__ void stage_async(bf16* dst, const bf16* src,
                                            size_t lda, int tid) {
#pragma unroll
    for (int c = 0; c < 2; ++c) {
        int chunk = tid * 2 + c;            // 0..511
        int row   = chunk >> 2;             // 0..127
        int col   = (chunk & 3) * 8;        // 0,8,16,24
        async_copy_b128(dst + row * TILE_K + col,
                        src + (size_t)row * lda + col);
    }
}

// ---------------------------------------------------------------------------
// WMMA fragment loads per ISA layouts (16x16x32 bf16)
// ---------------------------------------------------------------------------
__device__ __forceinline__ v16bf cat16(v8bf lo, v8bf hi) {
    return __builtin_shufflevector(lo, hi, 0,1,2,3,4,5,6,7,8,9,10,11,12,13,14,15);
}

__device__ __forceinline__ v16bf load_fragA(const bf16* sA, int mBase, int lane) {
    int row  = mBase + (lane & 15);
    int koff = (lane >> 4) * 8;
    v8bf lo = *(const v8bf*)(sA + row * TILE_K + koff);
    v8bf hi = *(const v8bf*)(sA + row * TILE_K + koff + 16);
    return cat16(lo, hi);
}

__device__ __forceinline__ v16bf load_fragB(const bf16* sB, int nBase, int lane) {
    int row  = nBase + (lane & 15);
    int koff = (lane >> 4) * 8;
    v8bf lo = *(const v8bf*)(sB + row * TILE_K + koff);
    v8bf hi = *(const v8bf*)(sB + row * TILE_K + koff + 16);
    return cat16(lo, hi);
}

// one staged K-step: 8 waves, wave (warpM 0..3, warpN 0..1) owns 32x64 patch
__device__ __forceinline__ void mma_step(const bf16* sA, const bf16* sB,
                                         v8f acc[2][4], int lane, int warpM, int warpN) {
    v16bf afr[2], bfr[4];
#pragma unroll
    for (int i = 0; i < 2; ++i) afr[i] = load_fragA(sA, warpM * 32 + i * 16, lane);
#pragma unroll
    for (int j = 0; j < 4; ++j) bfr[j] = load_fragB(sB, warpN * 64 + j * 16, lane);
#pragma unroll
    for (int i = 0; i < 2; ++i)
#pragma unroll
        for (int j = 0; j < 4; ++j)
            acc[i][j] = __builtin_amdgcn_wmma_f32_16x16x32_bf16(
                false, afr[i], false, bfr[j], (short)0, acc[i][j], false, false);
}

// ---------------------------------------------------------------------------
// Kernel 1: projection  C[m,n] = sum_k A[m,k]*W[n,k] + bias[n]   (bf16 out)
//   TDM double-buffered staging, wave 0 is the DMA leader.
// ---------------------------------------------------------------------------
__global__ void __launch_bounds__(THREADS) proj_gemm(
        const bf16* __restrict__ A, const bf16* __restrict__ W,
        const float* __restrict__ bias, bf16* __restrict__ C,
        int K, int lda, int ldb, int ldc) {
    __shared__ alignas(16) bf16 sA[2 * LDS_TILE];
    __shared__ alignas(16) bf16 sB[2 * LDS_TILE];
    const int tid = threadIdx.x, lane = tid & 31, wave = tid >> 5;
    const int warpM = wave & 3, warpN = wave >> 2;
    const int mTile = blockIdx.y * TILE_M, nTile = blockIdx.x * TILE_N;
    const bf16* Ab = A + (size_t)mTile * lda;
    const bf16* Bb = W + (size_t)nTile * ldb;
    const bool leader = (wave == 0);
    v8f acc[2][4] = {};
    const int nIter = K / TILE_K;
    if (leader) {
        tdm_load_tile_2d(sA, Ab, TILE_K, TILE_M, (u64)lda);
        tdm_load_tile_2d(sB, Bb, TILE_K, TILE_N, (u64)ldb);
    }
    for (int t = 0; t < nIter; ++t) {
        const int buf = t & 1;
        if (t + 1 < nIter) {
            if (leader) {
                tdm_load_tile_2d(sA + ((t + 1) & 1) * LDS_TILE,
                                 Ab + (t + 1) * TILE_K, TILE_K, TILE_M, (u64)lda);
                tdm_load_tile_2d(sB + ((t + 1) & 1) * LDS_TILE,
                                 Bb + (t + 1) * TILE_K, TILE_K, TILE_N, (u64)ldb);
                WAIT_TENSOR(2);   // tile t landed; tile t+1 stays in flight
            }
        } else if (leader) {
            WAIT_TENSOR(0);
        }
        __syncthreads();
        mma_step(sA + buf * LDS_TILE, sB + buf * LDS_TILE, acc, lane, warpM, warpN);
        __syncthreads();
    }
    const int rowHalf = (lane >> 4) * 8, colL = lane & 15;
#pragma unroll
    for (int i = 0; i < 2; ++i)
#pragma unroll
        for (int j = 0; j < 4; ++j) {
            int m0 = mTile + warpM * 32 + i * 16 + rowHalf;
            int n0 = nTile + warpN * 64 + j * 16 + colL;
            float bb = bias[n0];
#pragma unroll
            for (int r = 0; r < 8; ++r)
                C[(size_t)(m0 + r) * ldc + n0] = (bf16)(acc[i][j][r] + bb);
        }
}

// ---------------------------------------------------------------------------
// Kernel 2: energy per (n,h):  E[q,k] = mask ? (Q_nh . K_nh^T)[q,k]/sqrt(E) : 0
// ---------------------------------------------------------------------------
__global__ void __launch_bounds__(THREADS) energy_gemm(
        const bf16* __restrict__ Qp, const bf16* __restrict__ Kp,
        const int* __restrict__ mask, bf16* __restrict__ En) {
    __shared__ alignas(16) bf16 sA[2 * LDS_TILE];
    __shared__ alignas(16) bf16 sB[2 * LDS_TILE];
    const int z = blockIdx.z, n = z >> 3, h = z & 7;
    const int tid = threadIdx.x, lane = tid & 31, wave = tid >> 5;
    const int warpM = wave & 3, warpN = wave >> 2;
    const int mTile = blockIdx.y * TILE_M, nTile = blockIdx.x * TILE_N;
    const bf16* Ab = Qp + (size_t)n * SEQ * EH + (size_t)h * EMB + (size_t)mTile * EH;
    const bf16* Bb = Kp + (size_t)n * SEQ * EH + (size_t)h * EMB + (size_t)nTile * EH;
    bf16* C = En + (size_t)z * SEQ * SEQ;
    const int* mrow = mask + (size_t)n * SEQ * SEQ;
    const bool leader = (wave == 0);
    v8f acc[2][4] = {};
    const int nIter = EMB / TILE_K;
    if (leader) {
        tdm_load_tile_2d(sA, Ab, TILE_K, TILE_M, EH);
        tdm_load_tile_2d(sB, Bb, TILE_K, TILE_N, EH);
    }
    for (int t = 0; t < nIter; ++t) {
        const int buf = t & 1;
        if (t + 1 < nIter) {
            if (leader) {
                tdm_load_tile_2d(sA + ((t + 1) & 1) * LDS_TILE,
                                 Ab + (t + 1) * TILE_K, TILE_K, TILE_M, EH);
                tdm_load_tile_2d(sB + ((t + 1) & 1) * LDS_TILE,
                                 Bb + (t + 1) * TILE_K, TILE_K, TILE_N, EH);
                WAIT_TENSOR(2);
            }
        } else if (leader) {
            WAIT_TENSOR(0);
        }
        __syncthreads();
        mma_step(sA + buf * LDS_TILE, sB + buf * LDS_TILE, acc, lane, warpM, warpN);
        __syncthreads();
    }
    const int rowHalf = (lane >> 4) * 8, colL = lane & 15;
#pragma unroll
    for (int i = 0; i < 2; ++i)
#pragma unroll
        for (int j = 0; j < 4; ++j) {
            int m0 = mTile + warpM * 32 + i * 16 + rowHalf;
            int n0 = nTile + warpN * 64 + j * 16 + colL;
#pragma unroll
            for (int r = 0; r < 8; ++r) {
                int q = m0 + r;
                int mv = mrow[(size_t)q * SEQ + n0];
                float e = mv ? acc[i][j][r] * INV_SQRT_E : 0.0f;
                C[(size_t)q * SEQ + n0] = (bf16)e;
            }
        }
}

// ---------------------------------------------------------------------------
// Kernel 3: out[n,q,d] = sum_h sum_l E[n,h,q,l] * V[n,l,h,d]
//   GEMM per batch: M=1024(q), N=512(d), K=8192(h*1024+l)
//   A staged via per-lane async DMA (ASYNCcnt); B register-transposed into LDS
// ---------------------------------------------------------------------------
__device__ __forceinline__ void stage_Bt(bf16* dst, const bf16* Bb,
                                         int h, int l0, int nTile, int tid) {
    const int klane = tid & 31;          // k row within staged tile
    const int dseg  = tid >> 5;          // 8 segments of 16 d-values
    const bf16* src = Bb + (size_t)(l0 + klane) * EH + (size_t)h * EMB
                      + nTile + dseg * 16;
    v8bf p0 = *(const v8bf*)(src);
    v8bf p1 = *(const v8bf*)(src + 8);
#pragma unroll
    for (int e = 0; e < 8; ++e) {
        dst[(dseg * 16 + e)     * TILE_K + klane] = p0[e];
        dst[(dseg * 16 + 8 + e) * TILE_K + klane] = p1[e];
    }
}

__global__ void __launch_bounds__(THREADS) attnv_gemm(
        const bf16* __restrict__ En, const bf16* __restrict__ Vp,
        bf16* __restrict__ At) {
    __shared__ alignas(16) bf16 sA[2 * LDS_TILE];
    __shared__ alignas(16) bf16 sB[2 * LDS_TILE];
    const int nb = blockIdx.z;
    const bf16* Ab = En + (size_t)nb * HDS * SEQ * SEQ;   // [h][q][l]
    const bf16* Bb = Vp + (size_t)nb * SEQ * EH;          // [l][h*512+d]
    const int tid = threadIdx.x, lane = tid & 31, wave = tid >> 5;
    const int warpM = wave & 3, warpN = wave >> 2;
    const int mTile = blockIdx.y * TILE_M, nTile = blockIdx.x * TILE_N;
    v8f acc[2][4] = {};
    const int nIter = (HDS * SEQ) / TILE_K;   // 256
    stage_async(sA, Ab + (size_t)mTile * SEQ, SEQ, tid);   // t=0: h=0, l0=0
    stage_Bt(sB, Bb, 0, 0, nTile, tid);
    for (int t = 0; t < nIter; ++t) {
        const int buf = t & 1;
        if (t + 1 < nIter) {
            int k1 = (t + 1) * TILE_K;
            int h = k1 >> 10, l0 = k1 & (SEQ - 1);   // TILE_K | SEQ: no h crossing
            stage_async(sA + ((t + 1) & 1) * LDS_TILE,
                        Ab + (size_t)h * SEQ * SEQ + (size_t)mTile * SEQ + l0, SEQ, tid);
            stage_Bt(sB + ((t + 1) & 1) * LDS_TILE, Bb, h, l0, nTile, tid);
            WAIT_ASYNC(0x2);      // only A uses ASYNCcnt here (2 ops/wave/tile)
        } else {
            WAIT_ASYNC(0x0);
        }
        __syncthreads();
        mma_step(sA + buf * LDS_TILE, sB + buf * LDS_TILE, acc, lane, warpM, warpN);
        __syncthreads();
    }
    const int rowHalf = (lane >> 4) * 8, colL = lane & 15;
#pragma unroll
    for (int i = 0; i < 2; ++i)
#pragma unroll
        for (int j = 0; j < 4; ++j) {
            int m0 = mTile + warpM * 32 + i * 16 + rowHalf;
            int n0 = nTile + warpN * 64 + j * 16 + colL;
#pragma unroll
            for (int r = 0; r < 8; ++r)
                At[((size_t)nb * SEQ + m0 + r) * EMB + n0] = (bf16)acc[i][j][r];
        }
}

// ---------------------------------------------------------------------------
// Kernel 4: final projection  out[m,n] = sum_k At[m,k]*Wo[n,k] + bo[n]  (f32)
// ---------------------------------------------------------------------------
__global__ void __launch_bounds__(THREADS) final_gemm(
        const bf16* __restrict__ A, const bf16* __restrict__ Wo,
        const float* __restrict__ bo, float* __restrict__ out) {
    __shared__ alignas(16) bf16 sA[2 * LDS_TILE];
    __shared__ alignas(16) bf16 sB[2 * LDS_TILE];
    const int tid = threadIdx.x, lane = tid & 31, wave = tid >> 5;
    const int warpM = wave & 3, warpN = wave >> 2;
    const int mTile = blockIdx.y * TILE_M, nTile = blockIdx.x * TILE_N;
    const bf16* Ab = A + (size_t)mTile * EMB;
    const bf16* Bb = Wo + (size_t)nTile * EMB;
    const bool leader = (wave == 0);
    v8f acc[2][4] = {};
    const int nIter = EMB / TILE_K;
    if (leader) {
        tdm_load_tile_2d(sA, Ab, TILE_K, TILE_M, EMB);
        tdm_load_tile_2d(sB, Bb, TILE_K, TILE_N, EMB);
    }
    for (int t = 0; t < nIter; ++t) {
        const int buf = t & 1;
        if (t + 1 < nIter) {
            if (leader) {
                tdm_load_tile_2d(sA + ((t + 1) & 1) * LDS_TILE,
                                 Ab + (t + 1) * TILE_K, TILE_K, TILE_M, EMB);
                tdm_load_tile_2d(sB + ((t + 1) & 1) * LDS_TILE,
                                 Bb + (t + 1) * TILE_K, TILE_K, TILE_N, EMB);
                WAIT_TENSOR(2);
            }
        } else if (leader) {
            WAIT_TENSOR(0);
        }
        __syncthreads();
        mma_step(sA + buf * LDS_TILE, sB + buf * LDS_TILE, acc, lane, warpM, warpN);
        __syncthreads();
    }
    const int rowHalf = (lane >> 4) * 8, colL = lane & 15;
#pragma unroll
    for (int i = 0; i < 2; ++i)
#pragma unroll
        for (int j = 0; j < 4; ++j) {
            int m0 = mTile + warpM * 32 + i * 16 + rowHalf;
            int n0 = nTile + warpN * 64 + j * 16 + colL;
            float bb = bo[n0];
#pragma unroll
            for (int r = 0; r < 8; ++r)
                out[(size_t)(m0 + r) * EMB + n0] = acc[i][j][r] + bb;
        }
}

// ---------------------------------------------------------------------------
extern "C" void kernel_launch(void* const* d_in, const int* in_sizes, int n_in,
                              void* d_out, int out_size, void* d_ws, size_t ws_size,
                              hipStream_t stream) {
    // input order: values, keys, query, pos, mask, Wv, bv, Wk, bk, Wq, bq, Wo, bo
    const float* f_values = (const float*)d_in[0];
    const float* f_keys   = (const float*)d_in[1];
    const float* f_query  = (const float*)d_in[2];
    const int*   mask     = (const int*)  d_in[4];
    const float* f_Wv     = (const float*)d_in[5];
    const float* bv       = (const float*)d_in[6];
    const float* f_Wk     = (const float*)d_in[7];
    const float* bk       = (const float*)d_in[8];
    const float* f_Wq     = (const float*)d_in[9];
    const float* bq       = (const float*)d_in[10];
    const float* f_Wo     = (const float*)d_in[11];
    const float* bo       = (const float*)d_in[12];
    float* out = (float*)d_out;

    char* ws = (char*)d_ws;
    bf16* vX = (bf16*)(ws + OFF_VX);
    bf16* kX = (bf16*)(ws + OFF_KX);
    bf16* qX = (bf16*)(ws + OFF_QX);
    bf16* wV = (bf16*)(ws + OFF_WV);
    bf16* wK = (bf16*)(ws + OFF_WK);
    bf16* wQ = (bf16*)(ws + OFF_WQ);
    bf16* wO = (bf16*)(ws + OFF_WO);
    bf16* Vp = (bf16*)(ws + OFF_VP);
    bf16* Kp = (bf16*)(ws + OFF_KP);
    bf16* Qp = (bf16*)(ws + OFF_QP);
    bf16* En = (bf16*)(ws + OFF_EN);
    bf16* At = (bf16*)(ws + OFF_AT);

    const int ACT_N8 = MROWS * EMB / 8;    // 524,288
    const int W_N8   = EH * EMB / 8;       // 262,144
    const int WO_N8  = EMB * EMB / 8;      //  32,768

    // 1) convert everything to bf16 (8-wide)
    cvt_f32_bf16<<<(ACT_N8 + 255) / 256, 256, 0, stream>>>(f_values, vX, ACT_N8);
    cvt_f32_bf16<<<(ACT_N8 + 255) / 256, 256, 0, stream>>>(f_keys,   kX, ACT_N8);
    cvt_f32_bf16<<<(ACT_N8 + 255) / 256, 256, 0, stream>>>(f_query,  qX, ACT_N8);
    cvt_f32_bf16<<<(W_N8   + 255) / 256, 256, 0, stream>>>(f_Wv, wV, W_N8);
    cvt_f32_bf16<<<(W_N8   + 255) / 256, 256, 0, stream>>>(f_Wk, wK, W_N8);
    cvt_f32_bf16<<<(W_N8   + 255) / 256, 256, 0, stream>>>(f_Wq, wQ, W_N8);
    cvt_f32_bf16<<<(WO_N8  + 255) / 256, 256, 0, stream>>>(f_Wo, wO, WO_N8);

    // 2) QKV projections: (8192x512) @ (4096x512)^T
    dim3 gProj(EH / TILE_N, MROWS / TILE_M);           // (32, 64)
    proj_gemm<<<gProj, THREADS, 0, stream>>>(vX, wV, bv, Vp, EMB, EMB, EMB, EH);
    proj_gemm<<<gProj, THREADS, 0, stream>>>(kX, wK, bk, Kp, EMB, EMB, EMB, EH);
    proj_gemm<<<gProj, THREADS, 0, stream>>>(qX, wQ, bq, Qp, EMB, EMB, EMB, EH);

    // 3) energy per (n,h) with mask + 1/sqrt(E)
    dim3 gEn(SEQ / TILE_N, SEQ / TILE_M, NB * HDS);    // (8, 8, 64)
    energy_gemm<<<gEn, THREADS, 0, stream>>>(Qp, Kp, mask, En);

    // 4) attention @ V, head-sum fused into the K=8192 reduction, per batch
    dim3 gAv(EMB / TILE_N, SEQ / TILE_M, NB);          // (4, 8, 8)
    attnv_gemm<<<gAv, THREADS, 0, stream>>>(En, Vp, At);

    // 5) output projection + bias -> f32 out
    dim3 gFin(EMB / TILE_N, MROWS / TILE_M);           // (4, 64)
    final_gemm<<<gFin, THREADS, 0, stream>>>(At, wO, bo, out);
}
